// QLSTM_65481071401588
// MI455X (gfx1250) — matrix operand order repaired and checked by reading, not compiled, not run on hardware
//
#include <hip/hip_runtime.h>
#include <hip/hip_bf16.h>

// Problem dims (from reference): T=512, B=64, D=1024, H=1024
#define T_STEPS 512
#define BATCH   64
#define DIM     1024
#define HID     1024
#define KTOT    2048   // D + H
#define NG      4096   // 4 * H (gates fused along N, interleaved per hidden unit)

typedef __attribute__((ext_vector_type(16))) __bf16 bf16x16;
typedef __attribute__((ext_vector_type(8)))  __bf16 bf16x8;
typedef __attribute__((ext_vector_type(8)))  float  v8f;

union V16U { bf16x16 v; bf16x8 h[2]; };

// B-matrix fragment (16-bit B 32x16 layout): lane holds K=8h..8h+7 and
// K=16+8h..16+8h+7 of one N column -> two 16 B chunks from the transposed
// weight row.
__device__ __forceinline__ bf16x16 load_frag_b(const __bf16* p) {
  V16U u;
  u.h[0] = *(const bf16x8*)(p);
  u.h[1] = *(const bf16x8*)(p + 16);
  return u.v;
}

__device__ __forceinline__ float sigmoid_fast(float x) {
  return 1.0f / (1.0f + __expf(-x));
}
__device__ __forceinline__ float tanh_fast(float x) {
  x = fminf(fmaxf(x, -15.0f), 15.0f);
  float e = __expf(2.0f * x);
  return (e - 1.0f) / (e + 1.0f);
}

// ---------------------------------------------------------------------------
// One-time converters (re-run every kernel_launch; deterministic).
// Weight layout: WT[n][k], n = 4*j + gate (f=0,i=1,g=2,o=3), k in [0,2048).
// Source W* are row-major (D+H, H): element [k][j] at k*HID + j.
// ---------------------------------------------------------------------------
__global__ void convert_weights(const float* __restrict__ Wf,
                                const float* __restrict__ Wi,
                                const float* __restrict__ Wg,
                                const float* __restrict__ Wo,
                                __bf16* __restrict__ Wbf) {
  size_t idx = (size_t)blockIdx.x * blockDim.x + threadIdx.x;  // NG*KTOT total
  int n = (int)(idx >> 11);        // / KTOT
  int k = (int)(idx & (KTOT - 1));
  int g = n & 3;
  int j = n >> 2;
  const float* W = (g == 0) ? Wf : (g == 1) ? Wi : (g == 2) ? Wg : Wo;
  float v = W[(size_t)k * HID + j];
  Wbf[idx] = (__bf16)v;
}

__global__ void convert_x(const float* __restrict__ x, __bf16* __restrict__ xb) {
  size_t i = (size_t)blockIdx.x * blockDim.x + threadIdx.x;  // T*B*D total
  float v = __builtin_nontemporal_load(&x[i]);
  xb[i] = (__bf16)v;
}

__global__ void init_state(float* __restrict__ cx, __bf16* __restrict__ h0) {
  int i = blockIdx.x * blockDim.x + threadIdx.x;  // B*H total
  cx[i] = 0.0f;
  h0[i] = (__bf16)0.0f;
}

// ---------------------------------------------------------------------------
// One LSTM time step. Grid: 32 blocks x 128 threads (4 wave32 per block).
// Global wave gw in [0,128) owns N columns [gw*32, gw*32+32) of the fused
// 64 x 4096 gate pre-activation GEMM; that is all 4 gates for hidden units
// j in [gw*8, gw*8+8) across the full batch.
//
// The A operand (x||h, 64 x 2048 bf16) is identical for all 4 waves of a
// block, so it is staged per-block into LDS with the CDNA5 async path
// (global_load_async_to_lds_b128, ASYNCcnt), triple-buffered: one barrier
// per 32-wide K step, next tile fill overlapped with current tile's WMMAs.
// All ring addresses are precomputed; buffer selection is a rotating counter.
// ---------------------------------------------------------------------------
__global__ __launch_bounds__(128) void lstm_step(
    const __bf16* __restrict__ Wbf,     // [NG][KTOT] bf16, gate-interleaved W^T
    const __bf16* __restrict__ xt,      // [BATCH][DIM] bf16, inputs at time t
    const __bf16* __restrict__ hprev,   // [BATCH][HID] bf16
    __bf16*       __restrict__ hnext,   // [BATCH][HID] bf16
    float*        __restrict__ cx,      // [BATCH][HID] f32 cell state
    const float* __restrict__ bF, const float* __restrict__ bI,
    const float* __restrict__ bG, const float* __restrict__ bO,
    float* __restrict__ out_t,          // out + t*B*H
    float* __restrict__ outHx, float* __restrict__ outCx, int is_last) {

  const int tid  = threadIdx.x;
  const int lane = tid & 31;
  const int w    = tid >> 5;
  const int gw   = blockIdx.x * 4 + w;     // 0..127
  const int nbase = gw * 32;
  const int lhi  = lane >> 4;              // 0 (lanes 0-15) / 1 (lanes 16-31)
  const int llo  = lane & 15;
  const int ka_off = lhi << 3;             // A-frag K sub-offset: 0 or 8
  const int kb_off = lhi << 3;             // B-frag K sub-offset: 0 or 8

  // A tile ring: 3 buffers x 64 rows x 32 K (bf16), rows padded to 40 elems
  // (80 B: keeps 16 B chunk alignment, spreads LDS banks).
  __shared__ __align__(16) __bf16 asmem[3][64][40];
  __shared__ float smem[4][64][36];

  // ---- precomputed async-issue addressing (2 chunks per thread per tile) ----
  const int arow = tid >> 2;               // chunk0 row; chunk1 row = arow+32
  const int akof = (tid & 3) * 8;          // 16 B chunk within a row
  const __bf16* gx = xt    + (size_t)arow * DIM + akof;  // x-half source base
  const __bf16* gh = hprev + (size_t)arow * DIM + akof;  // h-half source base
  const unsigned l0b0 = (unsigned)(unsigned long long)&asmem[0][arow][akof];
  const unsigned l0b1 = (unsigned)(unsigned long long)&asmem[1][arow][akof];
  const unsigned l0b2 = (unsigned)(unsigned long long)&asmem[2][arow][akof];
  // chunk1 LDS dest = chunk0 + 32 rows * 80 B = +2560 B (baked into asm offset
  // via a second operand add; keep explicit adds cheap and uniform).

  // ---- precomputed A-fragment read bases (one per ring buffer) ----
  const __bf16* ra0 = &asmem[0][llo][ka_off];
  const __bf16* ra1 = &asmem[1][llo][ka_off];
  const __bf16* ra2 = &asmem[2][llo][ka_off];

  v8f acc[4][2];
#pragma unroll
  for (int mi = 0; mi < 4; ++mi)
#pragma unroll
    for (int ni = 0; ni < 2; ++ni)
      acc[mi][ni] = (v8f){};

  // Straight-line async fill of tile kt into ring buffer `buf`.
  auto issue_tile = [&](int kt, int buf) {
    const __bf16* p = ((kt < 32) ? gx : gh) + (kt & 31) * 32;
    unsigned l0 = (buf == 0) ? l0b0 : (buf == 1) ? l0b1 : l0b2;
    unsigned l1 = l0 + 32 * 80;            // +32 rows
    const __bf16* p1 = p + (size_t)32 * DIM;
    asm volatile("global_load_async_to_lds_b128 %0, %1, off"
                 :: "v"(l0), "v"(p) : "memory");
    asm volatile("global_load_async_to_lds_b128 %0, %1, off"
                 :: "v"(l1), "v"(p1) : "memory");
  };

  issue_tile(0, 0);

  int cur = 0;                             // ring buffer of tile kt
  for (int kt = 0; kt < 64; ++kt) {
    const int nxt = (cur == 2) ? 0 : cur + 1;
    if (kt < 63) {
      issue_tile(kt + 1, nxt);
      asm volatile("s_wait_asynccnt 0x2" ::: "memory");  // tile kt landed
    } else {
      asm volatile("s_wait_asynccnt 0x0" ::: "memory");
    }
    __syncthreads();   // all waves' chunks of tile kt visible (3-buf ring: safe)

    // A fragments from LDS (16-bit A 16x32 layout): two 16 B chunks per lane.
    const __bf16* abase = (cur == 0) ? ra0 : (cur == 1) ? ra1 : ra2;
    bf16x16 af[4];
#pragma unroll
    for (int mi = 0; mi < 4; ++mi) {
      const __bf16* ar = abase + mi * 16 * 40;   // +16 rows per M tile
      V16U u;
      u.h[0] = *(const bf16x8*)(ar);
      u.h[1] = *(const bf16x8*)(ar + 16);
      af[mi] = u.v;
    }

    // B fragments straight from (L2-resident) bf16 weights.
    const int k0 = kt * 32;
    bf16x16 bfrag[2];
#pragma unroll
    for (int ni = 0; ni < 2; ++ni) {
      const __bf16* bp = Wbf + (size_t)(nbase + ni * 16 + llo) * KTOT + k0 + kb_off;
      bfrag[ni] = load_frag_b(bp);
      __builtin_prefetch(bp + 32, 0, 3);   // next K slice -> global_prefetch_b8
    }

#pragma unroll
    for (int mi = 0; mi < 4; ++mi)
#pragma unroll
      for (int ni = 0; ni < 2; ++ni)
        acc[mi][ni] = __builtin_amdgcn_wmma_f32_16x16x32_bf16(
            false, af[mi], false, bfrag[ni], (short)0, acc[mi][ni], false, false);

    cur = nxt;
  }

  // Stage C tiles to LDS: per-wave 64(M) x 32(N) block, pad 36 (16B-aligned rows).
#pragma unroll
  for (int mi = 0; mi < 4; ++mi)
#pragma unroll
    for (int ni = 0; ni < 2; ++ni)
#pragma unroll
      for (int r = 0; r < 8; ++r)
        smem[w][mi * 16 + r + 8 * lhi][ni * 16 + llo] = acc[mi][ni][r];
  __syncthreads();

  // Fused gate epilogue: 64 batches x 8 hidden units per wave.
  const int jbase = gw * 8;
#pragma unroll 4
  for (int it = 0; it < 16; ++it) {
    int e  = it * 32 + lane;   // 0..511
    int b  = e >> 3;           // batch row 0..63
    int jj = e & 7;            // local hidden unit 0..7
    float f  = smem[w][b][4 * jj + 0];
    float i_ = smem[w][b][4 * jj + 1];
    float g  = smem[w][b][4 * jj + 2];
    float o  = smem[w][b][4 * jj + 3];
    int j = jbase + jj;
    float F = sigmoid_fast(f  + bF[j]);
    float I = sigmoid_fast(i_ + bI[j]);
    float G = tanh_fast(g + bG[j]);
    float O = sigmoid_fast(o + bO[j]);
    size_t idx = (size_t)b * HID + j;
    float c = F * cx[idx] + I * G;
    cx[idx] = c;
    float h = O * tanh_fast(c);
    __builtin_nontemporal_store(h, &out_t[idx]);   // never re-read on-chip
    hnext[idx] = (__bf16)h;                        // bf16 ring copy for t+1
    if (is_last) { outHx[idx] = h; outCx[idx] = c; }
  }
}

// ---------------------------------------------------------------------------
// Host side. Workspace layout (bytes):
//   [0, 16 MiB)            Wbf  bf16 [4096][2048]
//   [+, +64 MiB)           xbf  bf16 [T][B][D]
//   [+, +256 KiB)          hbf  bf16 ring [2][B][H]
//   [+, +256 KiB)          cx   f32 [B][H]
// Total ~80.5 MiB.
// ---------------------------------------------------------------------------
extern "C" void kernel_launch(void* const* d_in, const int* in_sizes, int n_in,
                              void* d_out, int out_size, void* d_ws, size_t ws_size,
                              hipStream_t stream) {
  (void)in_sizes; (void)n_in; (void)out_size; (void)ws_size;

  const float* x  = (const float*)d_in[0];
  const float* Wf = (const float*)d_in[1];
  const float* bF = (const float*)d_in[2];
  const float* Wi = (const float*)d_in[3];
  const float* bI = (const float*)d_in[4];
  const float* Wg = (const float*)d_in[5];
  const float* bG = (const float*)d_in[6];
  const float* Wo = (const float*)d_in[7];
  const float* bO = (const float*)d_in[8];
  float* out = (float*)d_out;

  char* ws = (char*)d_ws;
  const size_t WB_BYTES = (size_t)NG * KTOT * 2;             // 16 MiB
  const size_t XB_BYTES = (size_t)T_STEPS * BATCH * DIM * 2; // 64 MiB
  const size_t HB_BYTES = (size_t)2 * BATCH * HID * 2;       // 256 KiB
  __bf16* Wbf = (__bf16*)ws;
  __bf16* xb  = (__bf16*)(ws + WB_BYTES);
  __bf16* hbf = (__bf16*)(ws + WB_BYTES + XB_BYTES);
  float*  cx  = (float*)(ws + WB_BYTES + XB_BYTES + HB_BYTES);

  // One-time precompute (per launch).
  convert_weights<<<(NG * KTOT) / 256, 256, 0, stream>>>(Wf, Wi, Wg, Wo, Wbf);
  convert_x<<<((size_t)T_STEPS * BATCH * DIM) / 256, 256, 0, stream>>>(x, xb);
  init_state<<<(BATCH * HID) / 256, 256, 0, stream>>>(cx, hbf);

  float* outHx = out + (size_t)T_STEPS * BATCH * HID;
  float* outCx = outHx + (size_t)BATCH * HID;

  for (int t = 0; t < T_STEPS; ++t) {
    const __bf16* xt = xb + (size_t)t * BATCH * DIM;
    const __bf16* hp = hbf + (size_t)(t & 1) * BATCH * HID;
    __bf16*       hn = hbf + (size_t)((t + 1) & 1) * BATCH * HID;
    lstm_step<<<32, 128, 0, stream>>>(
        Wbf, xt, hp, hn, cx, bF, bI, bG, bO,
        out + (size_t)t * BATCH * HID, outHx, outCx,
        (t == T_STEPS - 1) ? 1 : 0);
  }
}
